// rnn_predictor_79242146611552
// MI455X (gfx1250) — matrix operand order repaired and checked by reading, not compiled, not run on hardware
//
#include <hip/hip_runtime.h>
#include <math.h>

// RNN predictor, fully fused: input projection GEMM + T=512 recurrent steps
// + on-the-fly readout dot product. bf16 WMMA (v_wmma_f32_16x16x32_bf16),
// f32 accumulate. B=512, T=512, I=64, H=128.
//
// Grid: 32 blocks (16 batch rows each), 256 threads = 8 wave32s.
// Wave w owns output columns n0 = 16*w .. 16*w+15.
// Double-buffered h/x tiles in LDS -> single barrier per timestep.

#define Bq 512
#define Tq 512
#define Iq 64
#define Hq 128

typedef __attribute__((ext_vector_type(16))) __bf16 v16bf;
typedef __attribute__((ext_vector_type(8)))  float  v8f;

union FragU {
    uint4 q[2];
    v16bf v;
};

// A-matrix (16x32 MxK, bf16) fragment from LDS, row-major, stride ld elems.
// Per CDNA5 ISA layout: lane(0-15)=row m holds K = kb+8*half + {0..7, 16..23}.
__device__ __forceinline__ v16bf load_frag_A(const __bf16* lds, int ld, int kb, int lane) {
    const int m    = lane & 15;
    const int half = lane >> 4;
    const __bf16* p = lds + m * ld + kb + 8 * half;
    FragU f;
    f.q[0] = *(const uint4*)(p);        // K = base+0..7   -> VGPRs 0-3
    f.q[1] = *(const uint4*)(p + 16);   // K = base+16..23 -> VGPRs 4-7
    return f.v;
}

// B-matrix (32x16 KxN, bf16) fragment where B(k,n) = W[n0+n][kb+k] (i.e. W^T).
// Lane(0-15)=col n holds K = kb+8*half + {0..7, 16..23}.
__device__ __forceinline__ v16bf load_frag_B(const __bf16* lds, int ld, int kb, int n0, int lane) {
    const int n    = n0 + (lane & 15);
    const int half = lane >> 4;
    const __bf16* p = lds + n * ld + kb + 8 * half;
    FragU f;
    f.q[0] = *(const uint4*)(p);        // K = base+0..7   -> VGPRs 0-3
    f.q[1] = *(const uint4*)(p + 16);   // K = base+16..23 -> VGPRs 4-7
    return f.v;
}

// Branchless tanh: no exec-mask divergence on the scan's critical path.
// tanh(x) = sign(x) * (1 - e) / (1 + e), e = exp(-2|x|) in (0,1] (no overflow).
__device__ __forceinline__ float fast_tanh(float v) {
    const float a = fabsf(v);
    const float e = __expf(-2.0f * a);
    const float r = (1.0f - e) * __builtin_amdgcn_rcpf(1.0f + e);
    return copysignf(r, v);
}

__global__ __launch_bounds__(256) void rnn_fused_kernel(
    const float* __restrict__ x,      // [B,T,I]
    const float* __restrict__ W_ih,   // [H,I]
    const float* __restrict__ W_hh,   // [H,H]
    const float* __restrict__ b_ih,   // [H]
    const float* __restrict__ b_hh,   // [H]
    const float* __restrict__ W_out,  // [T*H]
    const float* __restrict__ b_out,  // [1]
    float* __restrict__ out)          // [B]
{
    __shared__ alignas(16) __bf16 Whh_s[Hq * Hq];     // 32 KB
    __shared__ alignas(16) __bf16 Wih_s[Hq * Iq];     // 16 KB
    __shared__ alignas(16) __bf16 h_s[2][16 * Hq];    // 2 x 4 KB (double buffer)
    __shared__ alignas(16) __bf16 x_s[2][16 * Iq];    // 2 x 2 KB (double buffer)
    __shared__ float bias_s[Hq];
    __shared__ float oacc_s[16];

    const int tid  = threadIdx.x;
    const int lane = tid & 31;
    const int wave = tid >> 5;         // 0..7
    const int n0   = wave * 16;        // this wave's N-tile
    const int b0   = blockIdx.x * 16;  // this block's batch rows

    // ---- one-time staging: weights f32 -> bf16 LDS, bias, h=0 ----
    for (int i = tid; i < Hq * Hq; i += 256) Whh_s[i] = (__bf16)W_hh[i];
    for (int i = tid; i < Hq * Iq; i += 256) Wih_s[i] = (__bf16)W_ih[i];
    for (int i = tid; i < Hq;      i += 256) bias_s[i] = b_ih[i] + b_hh[i];
    for (int i = tid; i < 16 * Hq; i += 256) h_s[0][i] = (__bf16)0.0f;
    if (tid < 16) oacc_s[tid] = 0.0f;

    // x tile load assignment (float4 per thread, 16 rows x 64 cols)
    const int xrow = tid >> 4;              // 0..15
    const int xcol = (tid & 15) * 4;        // 0..60 step 4
    const float* xbase = x + ((size_t)(b0 + xrow) * Tq) * Iq + xcol;
    const int xoff = xrow * Iq + xcol;

    // stage x tile for t=0 into buffer 0
    {
        const float4 xv = *(const float4*)(xbase);
        __bf16* d = &x_s[0][xoff];
        d[0] = (__bf16)xv.x; d[1] = (__bf16)xv.y;
        d[2] = (__bf16)xv.z; d[3] = (__bf16)xv.w;
    }
    __syncthreads();

    // ---- hoist B-fragments (loop-invariant) into registers ----
    v16bf Bih[2], Bhh[4];
#pragma unroll
    for (int f = 0; f < 2; ++f) Bih[f] = load_frag_B(Wih_s, Iq, 32 * f, n0, lane);
#pragma unroll
    for (int f = 0; f < 4; ++f) Bhh[f] = load_frag_B(Whh_s, Hq, 32 * f, n0, lane);

    const int   col   = n0 + (lane & 15);   // absolute N column of this lane
    const int   halfr = (lane >> 4) * 8;    // row offset for C/D layout
    const float biasv = bias_s[col];

    float acc[8];
#pragma unroll
    for (int r = 0; r < 8; ++r) acc[r] = 0.0f;

#pragma unroll 2
    for (int t = 0; t < Tq; ++t) {
        const int cur = t & 1;
        const int nxt = cur ^ 1;

        // software pipeline: issue next x tile + W_out loads first (hide latency)
        const int tp = (t + 1 < Tq) ? t + 1 : 0;   // clamped (last prefetch unused)
        const float4 xv = *(const float4*)(xbase + (size_t)tp * Iq);
        const float  wv = W_out[t * Hq + col];

        // issue ALL A-fragment loads, then WMMA (one dscnt wait, not six)
        v16bf Ax[2], Ah[4];
#pragma unroll
        for (int f = 0; f < 2; ++f) Ax[f] = load_frag_A(x_s[cur], Iq, 32 * f, lane);
#pragma unroll
        for (int f = 0; f < 4; ++f) Ah[f] = load_frag_A(h_s[cur], Hq, 32 * f, lane);

        // two independent accumulator chains -> shorter WMMA critical path
        v8f c0, c1;
#pragma unroll
        for (int r = 0; r < 8; ++r) { c0[r] = biasv; c1[r] = 0.0f; }
#pragma unroll
        for (int f = 0; f < 2; ++f)
            c0 = __builtin_amdgcn_wmma_f32_16x16x32_bf16(
                     false, Ax[f], false, Bih[f], (short)0, c0, false, false);
#pragma unroll
        for (int f = 0; f < 4; ++f)
            c1 = __builtin_amdgcn_wmma_f32_16x16x32_bf16(
                     false, Ah[f], false, Bhh[f], (short)0, c1, false, false);

        // h_new = tanh(c0+c1); fused readout accumulation; write next buffers
        float hn[8];
#pragma unroll
        for (int r = 0; r < 8; ++r) {
            hn[r] = fast_tanh(c0[r] + c1[r]);
            acc[r] += hn[r] * wv;
        }
#pragma unroll
        for (int r = 0; r < 8; ++r)
            h_s[nxt][(r + halfr) * Hq + col] = (__bf16)hn[r];
        {
            __bf16* d = &x_s[nxt][xoff];
            d[0] = (__bf16)xv.x; d[1] = (__bf16)xv.y;
            d[2] = (__bf16)xv.z; d[3] = (__bf16)xv.w;
        }

        __syncthreads();  // single barrier: next-step buffers published
    }

    // reduce readout: each lane's acc[r] belongs to batch row (r + halfr)
#pragma unroll
    for (int r = 0; r < 8; ++r)
        atomicAdd(&oacc_s[r + halfr], acc[r]);
    __syncthreads();

    if (tid < 16) {
        const float z = oacc_s[tid] + b_out[0];
        const float e = __expf(-z);
        out[b0 + tid] = __builtin_amdgcn_rcpf(1.0f + e);
    }
}

extern "C" void kernel_launch(void* const* d_in, const int* in_sizes, int n_in,
                              void* d_out, int out_size, void* d_ws, size_t ws_size,
                              hipStream_t stream) {
    (void)in_sizes; (void)n_in; (void)out_size; (void)d_ws; (void)ws_size;
    const float* x     = (const float*)d_in[0];
    const float* W_ih  = (const float*)d_in[1];
    const float* W_hh  = (const float*)d_in[2];
    const float* b_ih  = (const float*)d_in[3];
    const float* b_hh  = (const float*)d_in[4];
    const float* W_out = (const float*)d_in[5];
    const float* b_out = (const float*)d_in[6];
    float* out = (float*)d_out;

    rnn_fused_kernel<<<dim3(Bq / 16), dim3(256), 0, stream>>>(
        x, W_ih, W_hh, b_ih, b_hh, W_out, b_out, out);
}
